// AFNONet_85615878079217
// MI455X (gfx1250) — matrix operand order repaired and checked by reading, not compiled, not run on hardware
//
#include <hip/hip_runtime.h>

typedef _Float16 half_t;
typedef __attribute__((ext_vector_type(16))) _Float16 v16h;
typedef __attribute__((ext_vector_type(8)))  float    v8f;

#define B_      64
#define C_      3
#define IMG_    384
#define P_      16
#define GH_     24
#define D_      384
#define DEPTH_  12
#define KBLK_   8
#define BS_     48
#define WP_     13                 // rfft width 24 -> 13
#define NTOK_   (B_ * GH_ * GH_)   // 36864
#define M2_     (B_ * GH_ * WP_)   // 19968
#define MLPH_   (4 * D_)           // 1536
#define NCLS_   1000
#define KPATCH_ (C_ * P_ * P_)     // 768

static __device__ __forceinline__ float gelu_exact(float x) {
  return 0.5f * x * (1.0f + erff(x * 0.70710678118654752f));
}

// ---------------------------------------------------------------------------
// WMMA GEMM: C[M,N] = A[M,K](f16, row-major) * BT[N,K](f16, N-major) + bias
//  - double-buffered LDS tiles filled with GLOBAL_LOAD_ASYNC_TO_LDS_B64
//  - batched via blockIdx.z with element strides sA/sB/sBias/sOut
// ---------------------------------------------------------------------------
enum { EPI_BIAS = 0, EPI_POS = 1, EPI_RES = 2, EPI_RELU = 3, EPI_GELU = 4 };

template <int BM, int BN, int WR, int WC, int WMT, int WNT, int EPI>
__global__ __launch_bounds__(256) void gemm_wmma(
    const half_t* __restrict__ A, const half_t* __restrict__ BT,
    const float* __restrict__ bias, const float* extra,
    float* outF, half_t* outH,
    int M, int N, int Kd,
    long sA, long sB, long sBias, long sOut, int posMod)
{
  static_assert(WR * WC == 8, "8 waves per block");
  static_assert(BM == WR * WMT * 16 && BN == WC * WNT * 16, "tile mismatch");
  constexpr int KS = 32;
  constexpr int CA = (BM * KS) / (4 * 256);  // 8-byte chunks per thread (A)
  constexpr int CB = (BN * KS) / (4 * 256);  // 8-byte chunks per thread (B)
  constexpr int CHUNKS = CA + CB;            // async loads per thread per tile
  static_assert(CA >= 1 && CB >= 1, "chunking");

  __shared__ __align__(16) half_t As[2][BM][KS + 8];  // row pitch 80B (16B mult.)
  __shared__ __align__(16) half_t Bs[2][BN][KS + 8];

  const int bz = blockIdx.z;
  const half_t* Ab  = A  + (long)bz * sA;
  const half_t* BTb = BT + (long)bz * sB;
  const float*  biasb = bias + (long)bz * sBias;
  const long outOff = (long)bz * sOut;

  const int m0 = blockIdx.y * BM;
  const int n0 = blockIdx.x * BN;
  const int tid  = threadIdx.x;
  const int lane = tid & 31;
  const int wave = tid >> 5;
  const int mw = (wave / WC) * (WMT * 16);
  const int nw = (wave % WC) * (WNT * 16);
  const int nloc   = lane & 15;
  const int khalfA = (lane < 16) ? 0 : 8;   // A: K groups {0-7,16-23}/{8-15,24-31}
  const int kbaseB = (lane < 16) ? 0 : 8;   // B: K groups {0-7,16-23}/{8-15,24-31}

  // async-copy one KS-deep tile (A rows + BT rows are both K-contiguous)
  auto issue = [&](int buf, int k0) {
#pragma unroll
    for (int c = 0; c < CA; ++c) {
      const int idx = tid + c * 256;
      const int row = idx / (KS / 4);
      const int col = (idx % (KS / 4)) * 4;
      const unsigned int lds = (unsigned int)(unsigned long long)&As[buf][row][col];
      const half_t* g = Ab + (long)(m0 + row) * Kd + k0 + col;
      asm volatile("global_load_async_to_lds_b64 %0, %1, off"
                   :: "v"(lds), "v"(g) : "memory");
    }
#pragma unroll
    for (int c = 0; c < CB; ++c) {
      const int idx = tid + c * 256;
      const int row = idx / (KS / 4);
      const int col = (idx % (KS / 4)) * 4;
      const unsigned int lds = (unsigned int)(unsigned long long)&Bs[buf][row][col];
      const half_t* g = BTb + (long)(n0 + row) * Kd + k0 + col;
      asm volatile("global_load_async_to_lds_b64 %0, %1, off"
                   :: "v"(lds), "v"(g) : "memory");
    }
  };

  v8f acc[WMT][WNT];
  const v8f vzero = {0.f, 0.f, 0.f, 0.f, 0.f, 0.f, 0.f, 0.f};
  for (int i = 0; i < WMT; ++i)
    for (int j = 0; j < WNT; ++j) acc[i][j] = vzero;

  const int nT = Kd / KS;
  issue(0, 0);
  if (nT > 1) issue(1, KS);
  int cur = 0;

  for (int t = 0; t < nT; ++t) {
    if (t + 1 < nT)
      asm volatile("s_wait_asynccnt %0" :: "n"(CHUNKS) : "memory");  // tile t done, t+1 in flight
    else
      asm volatile("s_wait_asynccnt 0" ::: "memory");
    __syncthreads();

#pragma unroll
    for (int i = 0; i < WMT; ++i) {
      v16h a;
      const int ar = mw + i * 16 + nloc;  // M index = lane%16
#pragma unroll
      for (int t8 = 0; t8 < 8; ++t8) {
        a[t8]     = As[cur][ar][khalfA + t8];
        a[t8 + 8] = As[cur][ar][khalfA + 16 + t8];
      }
#pragma unroll
      for (int j = 0; j < WNT; ++j) {
        v16h b;
        const int br = nw + j * 16 + nloc;  // N index = lane%16
#pragma unroll
        for (int t16 = 0; t16 < 16; ++t16) b[t16] = Bs[cur][br][kbaseB + (t16 & 7) + ((t16 >> 3) << 4)];
        acc[i][j] = __builtin_amdgcn_wmma_f32_16x16x32_f16(
            false, a, false, b, (short)0, acc[i][j], false, false);
      }
    }
    __syncthreads();                       // everyone done reading buf[cur]
    if (t + 2 < nT) issue(cur, (t + 2) * KS);
    cur ^= 1;
  }

  const int rsub = (lane >> 4) << 3;  // 0 or 8
#pragma unroll
  for (int i = 0; i < WMT; ++i)
#pragma unroll
    for (int j = 0; j < WNT; ++j)
#pragma unroll
      for (int v = 0; v < 8; ++v) {
        const int row = m0 + mw + i * 16 + v + rsub;
        const int col = n0 + nw + j * 16 + nloc;
        float val = acc[i][j][v] + biasb[col];
        if (EPI == EPI_POS)  val += extra[(long)(row % posMod) * N + col];
        if (EPI == EPI_RES)  val += extra[(long)row * N + col];
        if (EPI == EPI_RELU) val = fmaxf(val, 0.0f);
        if (EPI == EPI_GELU) val = gelu_exact(val);
        const long o = outOff + (long)row * N + col;
        if (outF) outF[o] = val;
        if (outH) outH[o] = (half_t)val;
      }
}

// ---------------------------------------------------------------------------
// LayerNorm: one wave per token (D=384); optional f32 / f16 outputs
// ---------------------------------------------------------------------------
__global__ __launch_bounds__(256) void ln_kernel(
    const float* __restrict__ x, const float* __restrict__ g,
    const float* __restrict__ bta, float* outF, half_t* outH, int M, int Dd)
{
  const int token = blockIdx.x * 8 + (threadIdx.x >> 5);
  const int lane = threadIdx.x & 31;
  if (token >= M) return;
  const float* row = x + (long)token * Dd;
  float s = 0.f, s2 = 0.f;
  for (int d = lane; d < Dd; d += 32) { float v = row[d]; s += v; s2 += v * v; }
  for (int off = 16; off > 0; off >>= 1) {
    s  += __shfl_xor(s, off, 32);
    s2 += __shfl_xor(s2, off, 32);
  }
  const float mean = s / Dd;
  const float var  = s2 / Dd - mean * mean;
  const float inv  = rsqrtf(var + 1e-5f);
  for (int d = lane; d < Dd; d += 32) {
    const float v = (row[d] - mean) * inv * g[d] + bta[d];
    if (outF) outF[(long)token * Dd + d] = v;
    if (outH) outH[(long)token * Dd + d] = (half_t)v;
  }
}

// ---------------------------------------------------------------------------
// Prep kernels (weights are emitted N-major: BT[n][k])
// ---------------------------------------------------------------------------
__global__ void twiddle_kernel(float* tc, float* ts) {
  const int i = blockIdx.x * blockDim.x + threadIdx.x;
  if (i < 24 * 24) {
    const int k = i / 24, h = i % 24;
    const float a = 6.283185307179586f * (float)(k * h) / 24.0f;
    tc[i] = cosf(a);
    ts[i] = sinf(a);
  }
}

__global__ void conv_f16_kernel(const float* __restrict__ src, half_t* __restrict__ dst, int n) {
  const int i = blockIdx.x * blockDim.x + threadIdx.x;
  if (i < n) dst[i] = (half_t)src[i];
}

// (L,K,N) f32 -> (L,N,K) f16
__global__ void transpose_f16_kernel(const float* __restrict__ src, half_t* __restrict__ dst,
                                     int K, int N, int L) {
  const long i = (long)blockIdx.x * blockDim.x + threadIdx.x;
  const long total = (long)L * K * N;
  if (i >= total) return;
  const int k = (int)(i % K);
  long r = i / K;
  const int n = (int)(r % N);
  const int l = (int)(r / N);
  dst[i] = (half_t)src[((long)l * K + k) * N + n];
}

// (DEPTH,2,K,48,48) -> (DEPTH,K,96out,96in) f16:  transpose of [[w0,w1],[-w1,w0]]
__global__ void afno_combine_w_kernel(const float* __restrict__ w, half_t* __restrict__ Wt) {
  const int i = blockIdx.x * blockDim.x + threadIdx.x;
  const int total = DEPTH_ * KBLK_ * 96 * 96;
  if (i >= total) return;
  const int ji = i % 96;            // input row (fast: K-contiguous)
  int r = i / 96;
  const int jo = r % 96; r /= 96;   // output col
  const int k = r % KBLK_;
  const int l = r / KBLK_;
  const float* w0 = w + (((long)l * 2 + 0) * KBLK_ + k) * (BS_ * BS_);
  const float* w1 = w + (((long)l * 2 + 1) * KBLK_ + k) * (BS_ * BS_);
  const int a = ji % BS_, b = jo % BS_;
  float v;
  if (ji < BS_) v = (jo < BS_) ? w0[a * BS_ + b] : w1[a * BS_ + b];
  else          v = (jo < BS_) ? -w1[a * BS_ + b] : w0[a * BS_ + b];
  Wt[i] = (half_t)v;
}

// (DEPTH,2,K,48) -> (DEPTH,K,96)
__global__ void afno_combine_b_kernel(const float* __restrict__ b, float* __restrict__ bc) {
  const int i = blockIdx.x * blockDim.x + threadIdx.x;
  const int total = DEPTH_ * KBLK_ * 96;
  if (i >= total) return;
  const int j = i % 96;
  int r = i / 96;
  const int k = r % KBLK_;
  const int l = r / KBLK_;
  const int sel = (j < BS_) ? 0 : 1;
  bc[i] = b[(((long)l * 2 + sel) * KBLK_ + k) * BS_ + (j % BS_)];
}

// x(B,C,384,384) -> Apatch f16 (NTOK x 768), k = (c*16+p)*16+q
__global__ void im2col_kernel(const float* __restrict__ x, half_t* __restrict__ A) {
  const long i = (long)blockIdx.x * blockDim.x + threadIdx.x;
  const long total = (long)NTOK_ * KPATCH_;
  if (i >= total) return;
  const int k = (int)(i % KPATCH_);
  const long m = i / KPATCH_;
  const int c = k / (P_ * P_);
  const int p = (k / P_) % P_;
  const int q = k % P_;
  const int gw = (int)(m % GH_);
  const int gh = (int)((m / GH_) % GH_);
  const int b  = (int)(m / (GH_ * GH_));
  A[i] = (half_t)x[(((long)b * C_ + c) * IMG_ + gh * P_ + p) * IMG_ + gw * P_ + q];
}

// ---------------------------------------------------------------------------
// FFT stages (24-point DFTs, ortho => 1/sqrt(24) per stage)
// ---------------------------------------------------------------------------
#define RS24 0.20412414523193154f

__global__ __launch_bounds__(256) void fft_h_kernel(
    const float* __restrict__ x, const float* __restrict__ tc,
    const float* __restrict__ ts, float* __restrict__ Zr, float* __restrict__ Zi)
{
  __shared__ float c[576], s[576];
  for (int i = threadIdx.x; i < 576; i += 256) { c[i] = tc[i]; s[i] = ts[i]; }
  __syncthreads();
  const long idx = (long)blockIdx.x * 256 + threadIdx.x;
  const long total = (long)B_ * GH_ * GH_ * D_;
  if (idx >= total) return;
  const int d = (int)(idx % D_);
  long r = idx / D_;
  const int w  = (int)(r % GH_); r /= GH_;
  const int kh = (int)(r % GH_);
  const int b  = (int)(r / GH_);
  float zr = 0.f, zi = 0.f;
  for (int h = 0; h < 24; ++h) {
    const float v = x[(((long)b * GH_ + h) * GH_ + w) * D_ + d];
    zr += v * c[kh * 24 + h];
    zi -= v * s[kh * 24 + h];
  }
  Zr[idx] = zr * RS24;
  Zi[idx] = zi * RS24;
}

__global__ __launch_bounds__(256) void fft_w_kernel(
    const float* __restrict__ Zr, const float* __restrict__ Zi,
    const float* __restrict__ tc, const float* __restrict__ ts,
    float* __restrict__ Xr, float* __restrict__ Xi, half_t* __restrict__ Ablk)
{
  __shared__ float c[576], s[576];
  for (int i = threadIdx.x; i < 576; i += 256) { c[i] = tc[i]; s[i] = ts[i]; }
  __syncthreads();
  const long idx = (long)blockIdx.x * 256 + threadIdx.x;
  const long total = (long)B_ * GH_ * WP_ * D_;
  if (idx >= total) return;
  const int d = (int)(idx % D_);
  long r = idx / D_;
  const int kw = (int)(r % WP_); r /= WP_;
  const int kh = (int)(r % GH_);
  const int b  = (int)(r / GH_);
  float xr = 0.f, xi = 0.f;
  for (int w = 0; w < 24; ++w) {
    const long zi_ = (((long)b * GH_ + kh) * GH_ + w) * D_ + d;
    const float zr = Zr[zi_], zim = Zi[zi_];
    const float cc = c[kw * 24 + w], ss = s[kw * 24 + w];
    xr += zr * cc + zim * ss;   // (zr+i*zi)*(cc - i*ss)
    xi += zim * cc - zr * ss;
  }
  xr *= RS24; xi *= RS24;
  Xr[idx] = xr;
  Xi[idx] = xi;
  // pack f16 A for block GEMM: block k = d/48, col j (re) / 48+j (im)
  const int kb = d / BS_, j = d % BS_;
  const long m2 = ((long)b * GH_ + kh) * WP_ + kw;
  half_t* ap = Ablk + ((long)kb * M2_ + m2) * 96;
  ap[j]       = (half_t)xr;
  ap[BS_ + j] = (half_t)xi;
}

// soft-threshold + add frequency residual (in place on Xr/Xi)
__global__ __launch_bounds__(256) void shrink_kernel(
    const float* __restrict__ Y, float* __restrict__ Xr, float* __restrict__ Xi)
{
  const long idx = (long)blockIdx.x * 256 + threadIdx.x;
  const long total = (long)KBLK_ * M2_ * BS_;
  if (idx >= total) return;
  const int j = (int)(idx % BS_);
  long r = idx / BS_;
  const long m2 = r % M2_;
  const int  k  = (int)(r / M2_);
  const long base = ((long)k * M2_ + m2) * 96;
  const float yr = Y[base + j], yi = Y[base + BS_ + j];
  const float amp = sqrtf(yr * yr + yi * yi + 1e-12f);
  const float sh  = fmaxf(amp - 0.01f, 0.0f) / amp;
  const long f = m2 * D_ + k * BS_ + j;
  Xr[f] += yr * sh;
  Xi[f] += yi * sh;
}

__global__ __launch_bounds__(256) void ifft_h_kernel(
    const float* __restrict__ Fr, const float* __restrict__ Fi,
    const float* __restrict__ tc, const float* __restrict__ ts,
    float* __restrict__ Wr, float* __restrict__ Wi)
{
  __shared__ float c[576], s[576];
  for (int i = threadIdx.x; i < 576; i += 256) { c[i] = tc[i]; s[i] = ts[i]; }
  __syncthreads();
  const long idx = (long)blockIdx.x * 256 + threadIdx.x;
  const long total = (long)B_ * GH_ * WP_ * D_;
  if (idx >= total) return;
  const int d = (int)(idx % D_);
  long r = idx / D_;
  const int kw = (int)(r % WP_); r /= WP_;
  const int h  = (int)(r % GH_);
  const int b  = (int)(r / GH_);
  float wr = 0.f, wi = 0.f;
  for (int kh = 0; kh < 24; ++kh) {
    const long fi_ = (((long)b * GH_ + kh) * WP_ + kw) * D_ + d;
    const float fr = Fr[fi_], fim = Fi[fi_];
    const float cc = c[h * 24 + kh], ss = s[h * 24 + kh];
    wr += fr * cc - fim * ss;   // (fr+i*fi)*(cc + i*ss)
    wi += fim * cc + fr * ss;
  }
  Wr[idx] = wr * RS24;
  Wi[idx] = wi * RS24;
}

// irfft over kw (Hermitian) + residual add into h
__global__ __launch_bounds__(256) void ifft_w_res_kernel(
    const float* __restrict__ Wr, const float* __restrict__ Wi,
    const float* __restrict__ tc, const float* __restrict__ ts,
    float* __restrict__ h)
{
  __shared__ float c[576], s[576];
  for (int i = threadIdx.x; i < 576; i += 256) { c[i] = tc[i]; s[i] = ts[i]; }
  __syncthreads();
  const long idx = (long)blockIdx.x * 256 + threadIdx.x;
  const long total = (long)B_ * GH_ * GH_ * D_;
  if (idx >= total) return;
  const int d = (int)(idx % D_);
  long r = idx / D_;
  const int w  = (int)(r % GH_); r /= GH_;
  const int hh = (int)(r % GH_);
  const int b  = (int)(r / GH_);
  const long base = (((long)b * GH_ + hh) * WP_) * D_ + d;
  float acc = Wr[base];                                   // kw = 0
  const float nyq = Wr[base + 12L * D_];                  // kw = 12 (Nyquist)
  acc += (w & 1) ? -nyq : nyq;
  for (int kw = 1; kw <= 11; ++kw) {
    const float cc = c[kw * 24 + w], ss = s[kw * 24 + w];
    acc += 2.0f * (Wr[base + (long)kw * D_] * cc - Wi[base + (long)kw * D_] * ss);
  }
  h[idx] += acc * RS24;
}

// ---------------------------------------------------------------------------
// Final pool + head
// ---------------------------------------------------------------------------
__global__ void pool_kernel(const float* __restrict__ lnf, float* __restrict__ pooled) {
  const int i = blockIdx.x * blockDim.x + threadIdx.x;
  if (i >= B_ * D_) return;
  const int d = i % D_, b = i / D_;
  float s = 0.f;
  for (int t = 0; t < GH_ * GH_; ++t) s += lnf[((long)b * GH_ * GH_ + t) * D_ + d];
  pooled[i] = s / (float)(GH_ * GH_);
}

__global__ void head_kernel(const float* __restrict__ pooled,
                            const float* __restrict__ hw,
                            const float* __restrict__ hb, float* __restrict__ out) {
  const int i = blockIdx.x * blockDim.x + threadIdx.x;
  if (i >= B_ * NCLS_) return;
  const int n = i % NCLS_, b = i / NCLS_;
  float acc = hb[n];
  for (int d = 0; d < D_; ++d) acc += pooled[b * D_ + d] * hw[(long)d * NCLS_ + n];
  out[i] = acc;
}

// ---------------------------------------------------------------------------
// Host launcher
// ---------------------------------------------------------------------------
static inline long cdivl(long a, long b) { return (a + b - 1) / b; }

extern "C" void kernel_launch(void* const* d_in, const int* in_sizes, int n_in,
                              void* d_out, int out_size, void* d_ws, size_t ws_size,
                              hipStream_t stream) {
  (void)in_sizes; (void)n_in; (void)out_size; (void)ws_size;
  const float* x       = (const float*)d_in[0];
  const float* patch_w = (const float*)d_in[1];
  const float* patch_b = (const float*)d_in[2];
  const float* pos     = (const float*)d_in[3];
  const float* ln1_g   = (const float*)d_in[4];
  const float* ln1_b   = (const float*)d_in[5];
  const float* w1      = (const float*)d_in[6];
  const float* b1      = (const float*)d_in[7];
  const float* w2      = (const float*)d_in[8];
  const float* b2      = (const float*)d_in[9];
  const float* ln2_g   = (const float*)d_in[10];
  const float* ln2_b   = (const float*)d_in[11];
  const float* fc1_w   = (const float*)d_in[12];
  const float* fc1_b   = (const float*)d_in[13];
  const float* fc2_w   = (const float*)d_in[14];
  const float* fc2_b   = (const float*)d_in[15];
  const float* lnf_g   = (const float*)d_in[16];
  const float* lnf_b   = (const float*)d_in[17];
  const float* head_w  = (const float*)d_in[18];
  const float* head_b  = (const float*)d_in[19];
  float* out = (float*)d_out;

  char* ws = (char*)d_ws;
  size_t off = 0;
  auto alloc = [&](size_t bytes) -> void* {
    void* p = ws + off;
    off = (off + bytes + 255) & ~(size_t)255;
    return p;
  };

  half_t* Apatch = (half_t*)alloc((size_t)NTOK_ * KPATCH_ * 2);
  half_t* WpT    = (half_t*)alloc((size_t)D_ * KPATCH_ * 2);        // [N=384][K=768]
  half_t* fc1T   = (half_t*)alloc((size_t)DEPTH_ * MLPH_ * D_ * 2); // [N=1536][K=384]
  half_t* fc2T   = (half_t*)alloc((size_t)DEPTH_ * D_ * MLPH_ * 2); // [N=384][K=1536]
  half_t* W1cT   = (half_t*)alloc((size_t)DEPTH_ * KBLK_ * 96 * 96 * 2);
  half_t* W2cT   = (half_t*)alloc((size_t)DEPTH_ * KBLK_ * 96 * 96 * 2);
  float*  b1c    = (float*)alloc((size_t)DEPTH_ * KBLK_ * 96 * 4);
  float*  b2c    = (float*)alloc((size_t)DEPTH_ * KBLK_ * 96 * 4);
  float*  tc     = (float*)alloc(576 * 4);
  float*  tsn    = (float*)alloc(576 * 4);
  float*  hbuf   = (float*)alloc((size_t)NTOK_ * D_ * 4);
  float*  ln1buf = (float*)alloc((size_t)NTOK_ * D_ * 4);
  half_t* ln216  = (half_t*)alloc((size_t)NTOK_ * D_ * 2);
  half_t* mlph   = (half_t*)alloc((size_t)NTOK_ * MLPH_ * 2);
  float*  Zr     = (float*)alloc((size_t)NTOK_ * D_ * 4);
  float*  Zi     = (float*)alloc((size_t)NTOK_ * D_ * 4);
  float*  Xr     = (float*)alloc((size_t)M2_ * D_ * 4);
  float*  Xi     = (float*)alloc((size_t)M2_ * D_ * 4);
  half_t* Ablk   = (half_t*)alloc((size_t)KBLK_ * M2_ * 96 * 2);
  half_t* Hblk   = (half_t*)alloc((size_t)KBLK_ * M2_ * 96 * 2);
  float*  Yblk   = (float*)alloc((size_t)KBLK_ * M2_ * 96 * 4);
  float*  Wr2    = (float*)alloc((size_t)M2_ * D_ * 4);
  float*  Wi2    = (float*)alloc((size_t)M2_ * D_ * 4);
  float*  pooled = (float*)alloc((size_t)B_ * D_ * 4);

  const int T = 256;

  // ---- one-time prep (all B matrices emitted N-major) ----
  twiddle_kernel<<<cdivl(576, T), T, 0, stream>>>(tc, tsn);
  conv_f16_kernel<<<cdivl((long)D_ * KPATCH_, T), T, 0, stream>>>(patch_w, WpT, D_ * KPATCH_);
  transpose_f16_kernel<<<cdivl((long)DEPTH_ * D_ * MLPH_, T), T, 0, stream>>>(
      fc1_w, fc1T, D_, MLPH_, DEPTH_);
  transpose_f16_kernel<<<cdivl((long)DEPTH_ * MLPH_ * D_, T), T, 0, stream>>>(
      fc2_w, fc2T, MLPH_, D_, DEPTH_);
  afno_combine_w_kernel<<<cdivl((long)DEPTH_ * KBLK_ * 96 * 96, T), T, 0, stream>>>(w1, W1cT);
  afno_combine_w_kernel<<<cdivl((long)DEPTH_ * KBLK_ * 96 * 96, T), T, 0, stream>>>(w2, W2cT);
  afno_combine_b_kernel<<<cdivl((long)DEPTH_ * KBLK_ * 96, T), T, 0, stream>>>(b1, b1c);
  afno_combine_b_kernel<<<cdivl((long)DEPTH_ * KBLK_ * 96, T), T, 0, stream>>>(b2, b2c);

  // ---- patch embed: h = im2col(x) @ Wp + b + pos ----
  im2col_kernel<<<cdivl((long)NTOK_ * KPATCH_, T), T, 0, stream>>>(x, Apatch);
  {
    dim3 g(D_ / 64, NTOK_ / 128, 1);
    gemm_wmma<128, 64, 4, 2, 2, 2, EPI_POS><<<g, T, 0, stream>>>(
        Apatch, WpT, patch_b, pos, hbuf, nullptr,
        NTOK_, D_, KPATCH_, 0, 0, 0, 0, GH_ * GH_);
  }

  const long fhTot = (long)NTOK_ * D_;        // B*24*24*D
  const long fwTot = (long)M2_ * D_;          // B*24*13*D

  for (int l = 0; l < DEPTH_; ++l) {
    // --- AFNO mixer ---
    ln_kernel<<<cdivl(NTOK_, 8), T, 0, stream>>>(
        hbuf, ln1_g + l * D_, ln1_b + l * D_, ln1buf, nullptr, NTOK_, D_);
    fft_h_kernel<<<cdivl(fhTot, T), T, 0, stream>>>(ln1buf, tc, tsn, Zr, Zi);
    fft_w_kernel<<<cdivl(fwTot, T), T, 0, stream>>>(Zr, Zi, tc, tsn, Xr, Xi, Ablk);
    {
      dim3 g(96 / 32, M2_ / 128, KBLK_);
      gemm_wmma<128, 32, 4, 2, 2, 1, EPI_RELU><<<g, T, 0, stream>>>(
          Ablk, W1cT + (long)l * KBLK_ * 96 * 96, b1c + (long)l * KBLK_ * 96, nullptr,
          nullptr, Hblk, M2_, 96, 96,
          (long)M2_ * 96, 96L * 96, 96L, (long)M2_ * 96, 1);
      gemm_wmma<128, 32, 4, 2, 2, 1, EPI_BIAS><<<g, T, 0, stream>>>(
          Hblk, W2cT + (long)l * KBLK_ * 96 * 96, b2c + (long)l * KBLK_ * 96, nullptr,
          Yblk, nullptr, M2_, 96, 96,
          (long)M2_ * 96, 96L * 96, 96L, (long)M2_ * 96, 1);
    }
    shrink_kernel<<<cdivl((long)KBLK_ * M2_ * BS_, T), T, 0, stream>>>(Yblk, Xr, Xi);
    ifft_h_kernel<<<cdivl(fwTot, T), T, 0, stream>>>(Xr, Xi, tc, tsn, Wr2, Wi2);
    ifft_w_res_kernel<<<cdivl(fhTot, T), T, 0, stream>>>(Wr2, Wi2, tc, tsn, hbuf);

    // --- MLP ---
    ln_kernel<<<cdivl(NTOK_, 8), T, 0, stream>>>(
        hbuf, ln2_g + l * D_, ln2_b + l * D_, nullptr, ln216, NTOK_, D_);
    {
      dim3 g1(MLPH_ / 64, NTOK_ / 128, 1);
      gemm_wmma<128, 64, 4, 2, 2, 2, EPI_GELU><<<g1, T, 0, stream>>>(
          ln216, fc1T + (long)l * MLPH_ * D_, fc1_b + l * MLPH_, nullptr,
          nullptr, mlph, NTOK_, MLPH_, D_, 0, 0, 0, 0, 1);
      dim3 g2(D_ / 64, NTOK_ / 128, 1);
      gemm_wmma<128, 64, 4, 2, 2, 2, EPI_RES><<<g2, T, 0, stream>>>(
          mlph, fc2T + (long)l * D_ * MLPH_, fc2_b + l * D_, hbuf,
          hbuf, nullptr, NTOK_, D_, MLPH_, 0, 0, 0, 0, 1);
    }
  }

  // ---- final LN, pool, head ----
  ln_kernel<<<cdivl(NTOK_, 8), T, 0, stream>>>(hbuf, lnf_g, lnf_b, ln1buf, nullptr, NTOK_, D_);
  pool_kernel<<<cdivl((long)B_ * D_, T), T, 0, stream>>>(ln1buf, pooled);
  head_kernel<<<cdivl((long)B_ * NCLS_, T), T, 0, stream>>>(pooled, head_w, head_b, out);
}
